// LocalityAwareCrossAttention_35837207118362
// MI455X (gfx1250) — matrix operand order repaired and checked by reading, not compiled, not run on hardware
//
#include <hip/hip_runtime.h>

typedef unsigned short u16;
typedef unsigned int   u32;
typedef __attribute__((ext_vector_type(16))) __bf16 v16bf;
typedef __attribute__((ext_vector_type(8)))  float  v8f;

#define K_DIM   256
#define LDS_STR 264      // bf16 elems per LDS row (+8 pad -> 528B, conflict-free)
#define HGRID   56
#define NNB     29
#define NTOK    6272     // B*N = 2*3136

union FragU { uint4 u[2]; v16bf v; };

__device__ __forceinline__ u16 f32_to_bf16(float f) {
    u32 u = __float_as_uint(f);
    u = (u + 0x7FFFu + ((u >> 16) & 1u)) >> 16;   // round-to-nearest-even
    return (u16)u;
}

// ---------------------------------------------------------------------------
// Convert the four 256x256 weight matrices to bf16 (once per call).
// ---------------------------------------------------------------------------
__global__ __launch_bounds__(256) void cvt_weights(
    const float* __restrict__ wq, const float* __restrict__ wk,
    const float* __restrict__ wv, const float* __restrict__ wp,
    u16* __restrict__ oq, u16* __restrict__ ok,
    u16* __restrict__ ov, u16* __restrict__ op)
{
    int i = blockIdx.x * 256 + threadIdx.x;        // grid = 256 -> 65536 elems
    oq[i] = f32_to_bf16(wq[i]);
    ok[i] = f32_to_bf16(wk[i]);
    ov[i] = f32_to_bf16(wv[i]);
    op[i] = f32_to_bf16(wp[i]);
}

// ---------------------------------------------------------------------------
// y[m,n] = sum_k A[m,k] * W[n,k] (+ bias[n])   — bf16 WMMA 16x16x32, f32 acc.
// Block = 256 thr (8 waves). grid = (M/16, 2). Wave w -> column tile
// blockIdx.y*8+w. A tile (16x256) staged to LDS (converted f32->bf16 if
// A_IS_F32), W fragments read straight from the bf16 weight matrix (L2-hot).
// ---------------------------------------------------------------------------
template<bool A_IS_F32>
__global__ __launch_bounds__(256) void gemm_wmma(
    const void* __restrict__ Aptr,      // [M,256] f32 or bf16
    const u16*  __restrict__ Wbf,       // [256,256] bf16 row-major (row = out ch)
    const float* __restrict__ bias,     // nullptr or [256]
    float* __restrict__ Out,            // [M,256] f32
    int M)
{
    __shared__ u16 sA[16 * LDS_STR];
    const int tid  = threadIdx.x;
    const int wave = tid >> 5;
    const int lane = tid & 31;
    const long rowBase = (long)blockIdx.x * 16;
    const int  ctile   = blockIdx.y * 8 + wave;    // 0..15

    // ---- stage A tile (16 rows x 256 K) into LDS as bf16, coalesced ----
    if (A_IS_F32) {
        const float* A = (const float*)Aptr;
        #pragma unroll
        for (int i = 0; i < 16; ++i)
            sA[i * LDS_STR + tid] = f32_to_bf16(A[(rowBase + i) * K_DIM + tid]);
    } else {
        const u16* A = (const u16*)Aptr;
        #pragma unroll
        for (int i = 0; i < 16; ++i)
            sA[i * LDS_STR + tid] = A[(rowBase + i) * K_DIM + tid];
    }
    __syncthreads();

    const int m   = lane & 15;
    const int hlf = lane >> 4;
    const u16* aRow = sA  + m * LDS_STR;
    const u16* wRow = Wbf + (long)(ctile * 16 + m) * K_DIM;

    v8f acc = {};
    #pragma unroll
    for (int k0 = 0; k0 < K_DIM; k0 += 32) {
        FragU a, b;
        // A 16x32 bf16 layout: lane holds row m; K chunks at hlf*8 and 16+hlf*8
        const u16* ap = aRow + k0 + hlf * 8;
        a.u[0] = *(const uint4*)(ap);
        a.u[1] = *(const uint4*)(ap + 16);
        // B 32x16 bf16 layout: lane holds col n; K chunks at hlf*8 and 16+hlf*8
        const u16* bp = wRow + k0 + hlf * 8;
        b.u[0] = *(const uint4*)(bp);
        b.u[1] = *(const uint4*)(bp + 16);
        acc = __builtin_amdgcn_wmma_f32_16x16x32_bf16(
                  false, a.v, false, b.v, (short)0, acc, false, false);
    }

    // ---- epilogue: D[r] -> row rowBase + hlf*8 + r, col ctile*16 + m ----
    const int col = ctile * 16 + m;
    const float bv = bias ? bias[col] : 0.0f;
    float* outp = Out + (rowBase + hlf * 8) * K_DIM + col;
    #pragma unroll
    for (int r = 0; r < 8; ++r)
        outp[(long)r * K_DIM] = acc[r] + bv;
    (void)M;
}

// ---------------------------------------------------------------------------
// Radius-3 local attention: each query has <= 29 in-radius neighbors.
// Block = 256 thr = 8 waves = 8 heads, one block per token (b,n).
// Phase 1: lane j = neighbor -> dot(q, k_j) over 32 channels, wave softmax.
// Phase 2: lane c = channel  -> sum_j p_j * v[j][c]; emit bf16 for final GEMM.
// ---------------------------------------------------------------------------
__constant__ int NB_DY[NNB] = {-3,
    -2,-2,-2,-2,-2,  -1,-1,-1,-1,-1,
     0, 0, 0, 0, 0, 0, 0,
     1, 1, 1, 1, 1,   2, 2, 2, 2, 2,  3};
__constant__ int NB_DX[NNB] = { 0,
    -2,-1, 0, 1, 2,  -2,-1, 0, 1, 2,
    -3,-2,-1, 0, 1, 2, 3,
    -2,-1, 0, 1, 2,  -2,-1, 0, 1, 2,  0};

__global__ __launch_bounds__(256) void local_attn(
    const float* __restrict__ q, const float* __restrict__ k,
    const float* __restrict__ v, u16* __restrict__ outbf)
{
    __shared__ float qs[256];
    __shared__ float ps[8][32];
    __shared__ int   ms[8][32];

    const int bn   = blockIdx.x;                 // b*N + n
    const int n    = bn % (HGRID * HGRID);
    const int tid  = threadIdx.x;
    const int h    = tid >> 5;
    const int lane = tid & 31;
    const long rowBase = (long)bn * 256;
    const long batBase = (long)(bn - n) * 256;   // b*N*256

    qs[tid] = q[rowBase + tid];
    __syncthreads();

    const int y = n / HGRID, x = n % HGRID;
    float score = -INFINITY;
    int   mIdx  = -1;
    if (lane < NNB) {
        int ny = y + NB_DY[lane], nx = x + NB_DX[lane];
        if (ny >= 0 && ny < HGRID && nx >= 0 && nx < HGRID) {
            mIdx = ny * HGRID + nx;
            const float* kp = k + batBase + (long)mIdx * 256 + h * 32;
            const float* qp = &qs[h * 32];
            float d = 0.0f;
            #pragma unroll
            for (int c = 0; c < 32; c += 4) {
                float4 kv4 = *(const float4*)(kp + c);
                float4 qv4 = *(const float4*)(qp + c);
                d += kv4.x * qv4.x + kv4.y * qv4.y
                   + kv4.z * qv4.z + kv4.w * qv4.w;
            }
            score = d * 0.17677669529663689f;     // 1/sqrt(32)
        }
    }

    // wave32 softmax over the neighbor lanes
    float mx = score;
    #pragma unroll
    for (int off = 16; off; off >>= 1)
        mx = fmaxf(mx, __shfl_xor(mx, off, 32));
    float p = __expf(score - mx);                 // -inf lanes -> 0
    float s = p;
    #pragma unroll
    for (int off = 16; off; off >>= 1)
        s += __shfl_xor(s, off, 32);
    p = p / s;

    ps[h][lane] = p;
    ms[h][lane] = mIdx;
    __syncthreads();

    // Phase 2: lane = channel; v loads are 128B-coalesced across the wave
    float acc = 0.0f;
    const long vBase = batBase + h * 32 + lane;
    #pragma unroll
    for (int j = 0; j < NNB; ++j) {
        int mj = ms[h][j];
        if (mj >= 0)
            acc += ps[h][j] * v[vBase + (long)mj * 256];
    }
    outbf[rowBase + tid] = f32_to_bf16(acc);
}

// ---------------------------------------------------------------------------
extern "C" void kernel_launch(void* const* d_in, const int* in_sizes, int n_in,
                              void* d_out, int out_size, void* d_ws, size_t ws_size,
                              hipStream_t stream)
{
    const float* x   = (const float*)d_in[0];
    const float* xkv = (const float*)d_in[1];
    const float* Wq  = (const float*)d_in[2];
    const float* Wk  = (const float*)d_in[3];
    const float* Wv  = (const float*)d_in[4];
    const float* Wp  = (const float*)d_in[5];
    const float* bp  = (const float*)d_in[6];
    float* out = (float*)d_out;

    // workspace carve-up (~22.0 MB total)
    char* ws = (char*)d_ws;
    u16* wqb = (u16*)ws;                       // 4 x 65536 bf16 weights
    u16* wkb = wqb + 65536;
    u16* wvb = wkb + 65536;
    u16* wpb = wvb + 65536;
    float* qf = (float*)(ws + 4 * 65536 * sizeof(u16));
    float* kf = qf + (long)NTOK * K_DIM;
    float* vf = kf + (long)NTOK * K_DIM;
    u16* attnb = (u16*)(vf + (long)NTOK * K_DIM);

    cvt_weights<<<256, 256, 0, stream>>>(Wq, Wk, Wv, Wp, wqb, wkb, wvb, wpb);

    dim3 gg(NTOK / 16, 2), gb(256);
    gemm_wmma<true ><<<gg, gb, 0, stream>>>(x,     wqb, nullptr, qf, NTOK);
    gemm_wmma<true ><<<gg, gb, 0, stream>>>(xkv,   wkb, nullptr, kf, NTOK);
    gemm_wmma<true ><<<gg, gb, 0, stream>>>(xkv,   wvb, nullptr, vf, NTOK);

    local_attn<<<NTOK, 256, 0, stream>>>(qf, kf, vf, attnb);

    gemm_wmma<false><<<gg, gb, 0, stream>>>(attnb, wpb, bp, out, NTOK);

    (void)in_sizes; (void)n_in; (void)out_size; (void)ws_size;
}
